// BaselineModel_63848983822926
// MI455X (gfx1250) — compile-verified, matrix operand order checked by reading, not run on hardware
//
#include <hip/hip_runtime.h>
#include <hip/hip_bf16.h>

// ---------------------------------------------------------------------------
// HSTU attention block for MI455X (gfx1250, wave32, WMMA 16x16x32 f16->f32).
// B=8, S=1024, H=1024, NH=16, HD=64.
// - All matmuls (QKV proj, QK^T, PV, out proj) on v_wmma_f32_16x16x32_f16.
// - Attention K/V tiles staged by the Tensor Data Mover (tensor_load_to_lds,
//   TENSORcnt) with LDS row padding done by the TDM pad unit.
// - Output-GEMM A tiles staged with global_load_async_to_lds_b128 (ASYNCcnt).
// ---------------------------------------------------------------------------

#define DI __device__ __forceinline__

typedef __attribute__((ext_vector_type(16))) _Float16 v16h;
typedef __attribute__((ext_vector_type(8)))  _Float16 v8h;
typedef __attribute__((ext_vector_type(8)))  float    v8f;
typedef __attribute__((ext_vector_type(4)))  float    v4f;
typedef __attribute__((ext_vector_type(4)))  unsigned tdm_g0;   // D# group 0 (128b)
typedef __attribute__((ext_vector_type(8)))  unsigned tdm_g1;   // D# group 1 (256b)

constexpr int Bsz = 8, Ssz = 1024, Hsz = 1024, NHc = 16, HDc = 64;

// ---- workspace layout (float offsets) -------------------------------------
constexpr long OFF_W    = 0;                        // 4 softmax'd temporal weights
constexpr long OFF_COS  = 16;                       // S*HD rope cos
constexpr long OFF_SIN  = OFF_COS + (long)Ssz*HDc;  // S*HD rope sin
constexpr long XR_T     = (long)Bsz*NHc*Ssz*HDc;    // 8,388,608 per tensor
constexpr long OFF_XR   = OFF_SIN + (long)Ssz*HDc;  // 3 roped/biased f32 tensors
constexpr long POOL_T   = (long)Bsz*NHc*(512+256+128)*HDc; // 7,340,032 per tensor
constexpr long OFF_POOL = OFF_XR + 3*XR_T;          // pooled means, 3 tensors
constexpr long OFF_H    = OFF_POOL + 3*POOL_T;      // f16 region: QH | KH | VT
constexpr long OFF_ATTN = OFF_XR;                   // attention out overlays XR (consumed)
// LN f16 overlays OFF_POOL (consumed by then). total ws need ~= 240 MB.

constexpr long POOL_S2  = (long)Bsz*NHc*512*HDc;
constexpr long POOL_S3  = POOL_S2 + (long)Bsz*NHc*256*HDc;

DI long pool_scale_off(int scale) {
  return (scale == 1) ? 0 : (scale == 2 ? POOL_S2 : POOL_S3);
}

// ---- WMMA helpers ----------------------------------------------------------
// A fragment (16xK=32 f16): lane L holds row (L&15); elems 0..7 = K(half*8+0..7),
// elems 8..15 = K(16+half*8+0..7). B fragment is the mirrored layout with rows=N,
// so both load identically from row-major [16][32] tiles (stride%8==0 -> 16B aligned).
DI v16h load_frag16(const _Float16* base, int stride) {
  const int lane = threadIdx.x & 31;
  const int r    = lane & 15;
  const int half = lane >> 4;
  const _Float16* p = base + r * stride + half * 8;
  v8h lo = *(const v8h*)(p);
  v8h hi = *(const v8h*)(p + 16);
  return __builtin_shufflevector(lo, hi, 0,1,2,3,4,5,6,7,8,9,10,11,12,13,14,15);
}

DI v8f wmma_f16(v16h a, v16h b, v8f c) {
  return __builtin_amdgcn_wmma_f32_16x16x32_f16(false, a, false, b, (short)0, c,
                                                false, false);
}

// ---- TDM: 2-D tile load, global -> LDS (2-group descriptor form) ------------
// LDS byte address: flat shared pointers map addr[31:0] to the LDS offset
// (ISA 10.2 aperture rules), so truncating the flat address is the LDS address.
// pad: insert 4 DWORDs of padding every 32 DWORDs stored (row 128B -> LDS 144B),
//      i.e. pad_interval code 4, pad_amount code 3.
DI void tdm_load_2d(const void* lds_dst, const void* gsrc,
                    unsigned dim0_elems, unsigned dim1_rows,
                    unsigned tile0, unsigned tile1,
                    unsigned stride0_elems, bool pad) {
  unsigned long long ga = (unsigned long long)gsrc;
  unsigned lofs = (unsigned)(unsigned long long)lds_dst;
  tdm_g0 g0;
  g0[0] = 1u;                                          // count=1, user descriptor
  g0[1] = lofs;                                        // lds_addr (bytes)
  g0[2] = (unsigned)(ga & 0xffffffffu);                // global_addr[31:0]
  g0[3] = (unsigned)((ga >> 32) & 0x01ffffffu)         // global_addr[56:32]
          | (2u << 30);                                // type = 2 ("image")
  tdm_g1 g1;
  unsigned w0 = (1u << 16);                            // data_size = 1 (2-byte elems)
  if (pad) w0 |= (1u << 20) | (4u << 22) | (3u << 25); // pad_en, interval=32dw, amt=4dw
  g1[0] = w0;                                          // wg_mask=0 (not in cluster)
  g1[1] = (dim0_elems & 0xffffu) << 16;                // tensor_dim0[15:0] @ bit48
  g1[2] = ((dim0_elems >> 16) & 0xffffu)
          | ((dim1_rows & 0xffffu) << 16);             // dim0[31:16] | dim1[15:0]
  g1[3] = ((dim1_rows >> 16) & 0xffffu)
          | ((tile0 & 0xffffu) << 16);                 // dim1[31:16] | tile_dim0
  g1[4] = (tile1 & 0xffffu);                           // tile_dim1 (tile_dim2 = 0)
  g1[5] = stride0_elems;                               // tensor_dim0_stride[31:0]
  g1[6] = 0u;                                          // stride0[47:32] | dim1_stride lo
  g1[7] = 0u;
  asm volatile("tensor_load_to_lds %0, %1" :: "s"(g0), "s"(g1) : "memory");
}

// ---- Async global -> LDS 16B copy (no VGPR data path, ASYNCcnt) -------------
DI void async_copy_b128(const void* lds_dst, const void* gsrc) {
  unsigned lofs = (unsigned)(unsigned long long)lds_dst;
  asm volatile("global_load_async_to_lds_b128 %0, %1, off"
               :: "v"(lofs), "v"((unsigned long long)gsrc) : "memory");
}
DI void async_wait0() { asm volatile("s_wait_asynccnt 0x0" ::: "memory"); }

// ---------------------------------------------------------------------------
// Kernel 1: rope tables + temporal-weight softmax
// ---------------------------------------------------------------------------
__global__ __launch_bounds__(256) void prep_kernel(const float* tw, float* wsf) {
  int idx = blockIdx.x * 256 + threadIdx.x;          // 65536 = S*HD
  int s = idx >> 6, hd = idx & 63;
  float ang = (float)s * __expf(-(float)(hd & 62) * (9.2103403719761836f / 64.0f));
  float sn, cs;
  __sincosf(ang, &sn, &cs);
  wsf[OFF_COS + idx] = cs;
  wsf[OFF_SIN + idx] = sn;
  if (idx == 0) {
    float w0 = tw[0], w1 = tw[1], w2 = tw[2], w3 = tw[3];
    float m = fmaxf(fmaxf(w0, w1), fmaxf(w2, w3));
    float e0 = __expf(w0 - m), e1 = __expf(w1 - m), e2 = __expf(w2 - m), e3 = __expf(w3 - m);
    float inv = 1.0f / (e0 + e1 + e2 + e3);
    wsf[0] = e0 * inv; wsf[1] = e1 * inv; wsf[2] = e2 * inv; wsf[3] = e3 * inv;
  }
}

// ---------------------------------------------------------------------------
// Kernel 2: fused QKV projection GEMM + bias + RoPE + multi-scale pooling.
// Block tile = 128(M=seq) x 64(N=one full head). 8 waves, each 32x32 out tile.
// ---------------------------------------------------------------------------
__global__ __launch_bounds__(256) void qkv_gemm_kernel(
    const float* q, const float* k, const float* v,
    const float* Wq, const float* Wk, const float* Wv,
    const float* bq, const float* bk, const float* bv,
    float* wsf) {
  const int z = blockIdx.z;
  const float* X    = (z == 0) ? q  : (z == 1 ? k  : v);
  const float* W    = (z == 0) ? Wq : (z == 1 ? Wk : Wv);
  const float* bias = (z == 0) ? bq : (z == 1 ? bk : bv);

  const int tid   = threadIdx.x;
  const int m0    = blockIdx.y * 128;
  const int nh    = blockIdx.x;
  const int n0    = nh * 64;
  const int b     = m0 >> 10;
  const int sbase = m0 & 1023;
  const int wave  = tid >> 5, lane = tid & 31;
  const int wm = wave & 3, wn = wave >> 2;
  const int col = lane & 15, rowoff = (lane >> 4) * 8;

  // staging (f16 A[128][72] + B^T[64][72] = 27,648B) overlaid by D f32[128][68]
  __shared__ __align__(16) char smem[128 * 68 * 4];
  _Float16* At  = (_Float16*)smem;
  _Float16* Bt  = At + 128 * 72;
  float*    Dst = (float*)smem;

  v8f acc[2][2] = {};

  for (int k0 = 0; k0 < 1024; k0 += 64) {
#pragma unroll
    for (int i = 0; i < 8; ++i) {        // stage A: 128x64 f32 -> f16
      int idx = tid + i * 256;
      int row = idx >> 4, c4 = (idx & 15) * 4;
      v4f val = *(const v4f*)(X + (size_t)(m0 + row) * 1024 + k0 + c4);
      _Float16* d = At + row * 72 + c4;
      d[0] = (_Float16)val.x; d[1] = (_Float16)val.y;
      d[2] = (_Float16)val.z; d[3] = (_Float16)val.w;
    }
#pragma unroll
    for (int i = 0; i < 4; ++i) {        // stage W transposed: Bt[n][k]
      int idx = tid + i * 256;
      int kk = idx >> 4, c4 = (idx & 15) * 4;
      v4f val = *(const v4f*)(W + (size_t)(k0 + kk) * 1024 + n0 + c4);
      Bt[(c4 + 0) * 72 + kk] = (_Float16)val.x;
      Bt[(c4 + 1) * 72 + kk] = (_Float16)val.y;
      Bt[(c4 + 2) * 72 + kk] = (_Float16)val.z;
      Bt[(c4 + 3) * 72 + kk] = (_Float16)val.w;
    }
    if (k0 + 64 < 1024) {                // global_prefetch_b8 of next tiles
      __builtin_prefetch(X + (size_t)(m0 + (tid >> 1)) * 1024 + k0 + 64 + (tid & 1) * 32, 0, 1);
      __builtin_prefetch(W + (size_t)(k0 + 64 + (tid >> 2)) * 1024 + n0 + (tid & 3) * 16, 0, 1);
    }
    __syncthreads();
#pragma unroll
    for (int kk = 0; kk < 2; ++kk) {
      const int co = kk * 32;
      v16h a0 = load_frag16(At + (wm * 32) * 72 + co, 72);
      v16h a1 = load_frag16(At + (wm * 32 + 16) * 72 + co, 72);
      v16h b0 = load_frag16(Bt + (wn * 32) * 72 + co, 72);
      v16h b1 = load_frag16(Bt + (wn * 32 + 16) * 72 + co, 72);
      acc[0][0] = wmma_f16(a0, b0, acc[0][0]);
      acc[0][1] = wmma_f16(a0, b1, acc[0][1]);
      acc[1][0] = wmma_f16(a1, b0, acc[1][0]);
      acc[1][1] = wmma_f16(a1, b1, acc[1][1]);
    }
    __syncthreads();
  }

  // dump accumulators into LDS (C layout: vgpr r -> row r+rowoff, lane&15 -> col)
#pragma unroll
  for (int mm = 0; mm < 2; ++mm)
#pragma unroll
    for (int nn = 0; nn < 2; ++nn)
#pragma unroll
      for (int r = 0; r < 8; ++r)
        Dst[(wm * 32 + mm * 16 + rowoff + r) * 68 + wn * 32 + nn * 16 + col] = acc[mm][nn][r];
  __syncthreads();

  // bias + RoPE (pairs hd, hd+32 are block-local; one thread owns each pair)
  const float* cosT = wsf + OFF_COS;
  const float* sinT = wsf + OFF_SIN;
  float* XR = wsf + OFF_XR + (long)z * XR_T + ((long)(b * NHc + nh) * Ssz + sbase) * 64;
#pragma unroll
  for (int i = 0; i < 16; ++i) {
    int idx = tid + i * 256;             // 128 rows x 32 pairs
    int srow = idx >> 5, hd = idx & 31;
    int sg = sbase + srow;
    float vlo = Dst[srow * 68 + hd]      + bias[n0 + hd];
    float vhi = Dst[srow * 68 + hd + 32] + bias[n0 + hd + 32];
    float olo, ohi;
    if (z < 2) {
      float c0 = cosT[sg * 64 + hd],      s0 = sinT[sg * 64 + hd];
      float c1 = cosT[sg * 64 + hd + 32], s1 = sinT[sg * 64 + hd + 32];
      olo = vlo * c0 - vhi * s0;
      ohi = vhi * c1 + vlo * s1;
    } else { olo = vlo; ohi = vhi; }
    Dst[srow * 68 + hd] = olo;
    Dst[srow * 68 + hd + 32] = ohi;
    XR[(long)srow * 64 + hd]      = olo;
    XR[(long)srow * 64 + hd + 32] = ohi;
  }
  __syncthreads();

  // multi-scale pooled means (k=2,4,8 groups all inside this 128-row block)
  for (int idx = tid; idx < 112 * 64; idx += 256) {
    int g = idx >> 6, hd = idx & 63;
    int scale, kk, gg;
    if (g < 64)      { scale = 1; kk = 2; gg = g; }
    else if (g < 96) { scale = 2; kk = 4; gg = g - 64; }
    else             { scale = 3; kk = 8; gg = g - 96; }
    float s = 0.f;
    for (int j = 0; j < kk; ++j) s += Dst[(gg * kk + j) * 68 + hd];
    s *= (1.0f / (float)kk);
    int p = 1024 >> scale;
    wsf[OFF_POOL + (long)z * POOL_T + pool_scale_off(scale) +
        ((long)(b * NHc + nh) * p + (sbase >> scale) + gg) * 64 + hd] = s;
  }
}

// ---------------------------------------------------------------------------
// Kernel 3: temporal combine -> f16 Q, f16 K (row-major), f16 V^T (via LDS).
// ---------------------------------------------------------------------------
__global__ __launch_bounds__(256) void combine_kernel(float* wsf) {
  const int z  = blockIdx.z;
  const int bn = blockIdx.y;             // b*NH + nh
  const int s0 = blockIdx.x * 64;
  const int tid = threadIdx.x;
  const float wsc[4] = { wsf[0], wsf[1], wsf[2], wsf[3] };
  _Float16* HB = (_Float16*)(wsf + OFF_H);
  _Float16* QH = HB;
  _Float16* KH = HB + XR_T;
  _Float16* VT = HB + 2 * XR_T;
  __shared__ _Float16 tile[64 * 65];

  const float* XR    = wsf + OFF_XR + (long)z * XR_T + (long)bn * Ssz * 64;
  const float* PBase = wsf + OFF_POOL + (long)z * POOL_T;

#pragma unroll
  for (int i = 0; i < 16; ++i) {
    int idx = tid + i * 256;
    int srow = idx >> 6, hd = idx & 63;
    int s = s0 + srow;
    float acc = wsc[0] * XR[(long)s * 64 + hd];
#pragma unroll
    for (int scale = 1; scale <= 3; ++scale) {
      int kk = 1 << scale, p = 1024 >> scale;
      float coord = ((float)s + 0.5f) / (float)kk - 0.5f;
      coord = fmaxf(coord, 0.0f);
      int i0 = (int)coord; if (i0 > p - 1) i0 = p - 1;
      int i1 = i0 + 1;     if (i1 > p - 1) i1 = p - 1;
      float lam = coord - (float)i0;
      const float* PB = PBase + pool_scale_off(scale) + (long)bn * p * 64;
      float v0 = PB[(long)i0 * 64 + hd], v1 = PB[(long)i1 * 64 + hd];
      acc += wsc[scale] * (v0 + (v1 - v0) * lam);
    }
    _Float16 hv = (_Float16)acc;
    if (z == 0)      QH[((long)bn * Ssz + s) * 64 + hd] = hv;
    else if (z == 1) KH[((long)bn * Ssz + s) * 64 + hd] = hv;
    else             tile[hd * 65 + srow] = hv;     // stage for transpose
  }
  __syncthreads();
  if (z == 2) {
#pragma unroll
    for (int i = 0; i < 16; ++i) {
      int idx = tid + i * 256;
      int hd = idx >> 6, sc = idx & 63;
      VT[((long)bn * 64 + hd) * Ssz + s0 + sc] = tile[hd * 65 + sc];
    }
  }
}

// ---------------------------------------------------------------------------
// Kernel 4: flash attention. Block = (head, 64 q-rows), 4 waves x 16 q-rows.
// K/V tiles staged by the Tensor Data Mover (one wave issues, TENSORcnt+barrier).
// ---------------------------------------------------------------------------
__global__ __launch_bounds__(128) void attention_kernel(float* wsf) {
  const int bn  = blockIdx.y;
  const int q0  = blockIdx.x * 64;
  const int tid = threadIdx.x;
  const int wave = tid >> 5, lane = tid & 31;
  const int col = lane & 15, rowoff = (lane >> 4) * 8;

  _Float16* HB = (_Float16*)(wsf + OFF_H);
  const _Float16* QH = HB;
  const _Float16* KH = HB + XR_T;
  const _Float16* VT = HB + 2 * XR_T;
  float* ATTN = wsf + OFF_ATTN;

  __shared__ _Float16 Kt[64 * 72];        // keys x hd   (row 128B + 16B TDM pad)
  __shared__ _Float16 Vt[64 * 72];        // hd x keys   (V^T block, same pad)
  __shared__ _Float16 Pt[4 * 16 * 72];    // per-wave P tile

  v16h aq[2];
  {
    const _Float16* qp = QH + ((long)bn * Ssz + q0 + wave * 16) * 64;
    aq[0] = load_frag16(qp, 64);
    aq[1] = load_frag16(qp + 32, 64);
  }
  v8f o[4] = {};
  float mrun[8], lrun[8];
#pragma unroll
  for (int r = 0; r < 8; ++r) { mrun[r] = -1e30f; lrun[r] = 0.f; }

  for (int kb = 0; kb < 16; ++kb) {
    const int kbase = kb * 64;
    if (wave == 0) {
      // K block: 64 key rows x 64 halves, row stride 64 elems
      tdm_load_2d(Kt, KH + ((long)bn * Ssz + kbase) * 64,
                  /*dim0*/64, /*dim1*/64, /*tile0*/64, /*tile1*/64,
                  /*stride0*/64, /*pad*/true);
      // V^T block: 64 hd rows x 64 keys, row stride 1024 elems
      tdm_load_2d(Vt, VT + (long)bn * 64 * Ssz + kbase,
                  /*dim0*/64, /*dim1*/64, /*tile0*/64, /*tile1*/64,
                  /*stride0*/1024, /*pad*/true);
    }
    __builtin_amdgcn_s_wait_tensorcnt(0);
    __syncthreads();

    // scores: 16q x 64k per wave, 2 WMMAs (hd halves) per 16-key tile
    v8f sc[4] = {};
#pragma unroll
    for (int kt = 0; kt < 4; ++kt)
#pragma unroll
      for (int kk = 0; kk < 2; ++kk) {
        v16h bkf = load_frag16(Kt + (kt * 16) * 72 + kk * 32, 72);
        sc[kt] = wmma_f16(aq[kk], bkf, sc[kt]);
      }

    // online softmax per row (row = vgpr r + rowoff; N spread over 16-lane half)
    float alpha[8];
#pragma unroll
    for (int r = 0; r < 8; ++r) {
      float mx = -1e30f;
#pragma unroll
      for (int kt = 0; kt < 4; ++kt) { sc[kt][r] *= 0.125f; mx = fmaxf(mx, sc[kt][r]); }
#pragma unroll
      for (int off = 1; off < 16; off <<= 1) mx = fmaxf(mx, __shfl_xor(mx, off, 32));
      float nm = fmaxf(mrun[r], mx);
      alpha[r] = __expf(mrun[r] - nm);
      mrun[r] = nm;
      float rs = 0.f;
#pragma unroll
      for (int kt = 0; kt < 4; ++kt) {
        float pv = __expf(sc[kt][r] - nm);
        sc[kt][r] = pv;
        rs += pv;
      }
#pragma unroll
      for (int off = 1; off < 16; off <<= 1) rs += __shfl_xor(rs, off, 32);
      lrun[r] = lrun[r] * alpha[r] + rs;
    }
#pragma unroll
    for (int ht = 0; ht < 4; ++ht)
#pragma unroll
      for (int r = 0; r < 8; ++r) o[ht][r] *= alpha[r];

    // C-layout P -> LDS -> A-fragment relayout
    _Float16* pt = Pt + wave * 16 * 72;
#pragma unroll
    for (int kt = 0; kt < 4; ++kt)
#pragma unroll
      for (int r = 0; r < 8; ++r)
        pt[(rowoff + r) * 72 + kt * 16 + col] = (_Float16)sc[kt][r];
    __syncthreads();

    // O += P(16x64keys) * V(64keys x 64hd)
#pragma unroll
    for (int kk = 0; kk < 2; ++kk) {
      v16h ap = load_frag16(pt + kk * 32, 72);
#pragma unroll
      for (int ht = 0; ht < 4; ++ht) {
        v16h bv = load_frag16(Vt + (ht * 16) * 72 + kk * 32, 72);
        o[ht] = wmma_f16(ap, bv, o[ht]);
      }
    }
    __syncthreads();
  }

  const int b = bn >> 4, nh = bn & 15;
#pragma unroll
  for (int r = 0; r < 8; ++r) {
    float inv = 1.0f / lrun[r];
    int sg = q0 + wave * 16 + rowoff + r;
#pragma unroll
    for (int ht = 0; ht < 4; ++ht)
      ATTN[((long)b * Ssz + sg) * 1024 + nh * 64 + ht * 16 + col] = o[ht][r] * inv;
  }
}

// ---------------------------------------------------------------------------
// Kernel 5: LayerNorm over H per (b,s) row -> f16 (overlays pool buffer).
// ---------------------------------------------------------------------------
__global__ __launch_bounds__(256) void ln_kernel(const float* gamma, const float* beta,
                                                 float* wsf) {
  const int row = blockIdx.x;
  const int tid = threadIdx.x;
  const float* x = wsf + OFF_ATTN + (long)row * 1024;
  _Float16* LNH = (_Float16*)(wsf + OFF_POOL);

  float vv[4], s1 = 0.f, s2 = 0.f;
#pragma unroll
  for (int i = 0; i < 4; ++i) {
    vv[i] = x[tid + i * 256];
    s1 += vv[i];
    s2 += vv[i] * vv[i];
  }
#pragma unroll
  for (int off = 1; off < 32; off <<= 1) {
    s1 += __shfl_xor(s1, off, 32);
    s2 += __shfl_xor(s2, off, 32);
  }
  __shared__ float a1[8], a2[8];
  __shared__ float mu_s, rs_s;
  if ((tid & 31) == 0) { a1[tid >> 5] = s1; a2[tid >> 5] = s2; }
  __syncthreads();
  if (tid == 0) {
    float t1 = 0.f, t2 = 0.f;
    for (int i = 0; i < 8; ++i) { t1 += a1[i]; t2 += a2[i]; }
    float mu = t1 * (1.0f / 1024.0f);
    float var = t2 * (1.0f / 1024.0f) - mu * mu;
    mu_s = mu;
    rs_s = rsqrtf(var + 1e-5f);
  }
  __syncthreads();
  float mu = mu_s, rs = rs_s;
#pragma unroll
  for (int i = 0; i < 4; ++i) {
    int h = tid + i * 256;
    float y = (vv[i] - mu) * rs * gamma[h] + beta[h];
    LNH[(long)row * 1024 + h] = (_Float16)y;
  }
}

// ---------------------------------------------------------------------------
// Kernel 6: output projection GEMM -> d_out f32.
// A tiles are already f16: staged with global_load_async_to_lds_b128.
// ---------------------------------------------------------------------------
__global__ __launch_bounds__(256) void out_gemm_kernel(const float* Wo, const float* bo,
                                                       float* wsf, float* out) {
  const _Float16* LNH = (const _Float16*)(wsf + OFF_POOL);
  const int tid = threadIdx.x;
  const int m0 = blockIdx.y * 128;
  const int n0 = blockIdx.x * 64;
  const int wave = tid >> 5, lane = tid & 31;
  const int wm = wave & 3, wn = wave >> 2;
  const int col = lane & 15, rowoff = (lane >> 4) * 8;

  __shared__ _Float16 At[128 * 72];
  __shared__ _Float16 Bt[64 * 72];

  v8f acc[2][2] = {};

  for (int k0 = 0; k0 < 1024; k0 += 64) {
#pragma unroll
    for (int i = 0; i < 4; ++i) {        // A: 1024 16B chunks via async LDS DMA
      int idx = tid + i * 256;
      int row = idx >> 3, c8 = (idx & 7) * 8;
      async_copy_b128(At + row * 72 + c8,
                      LNH + (long)(m0 + row) * 1024 + k0 + c8);
    }
#pragma unroll
    for (int i = 0; i < 4; ++i) {        // Wo f32 -> f16 transposed
      int idx = tid + i * 256;
      int kk = idx >> 4, c4 = (idx & 15) * 4;
      v4f val = *(const v4f*)(Wo + (size_t)(k0 + kk) * 1024 + n0 + c4);
      Bt[(c4 + 0) * 72 + kk] = (_Float16)val.x;
      Bt[(c4 + 1) * 72 + kk] = (_Float16)val.y;
      Bt[(c4 + 2) * 72 + kk] = (_Float16)val.z;
      Bt[(c4 + 3) * 72 + kk] = (_Float16)val.w;
    }
    if (k0 + 64 < 1024)
      __builtin_prefetch(Wo + (size_t)(k0 + 64 + (tid >> 2)) * 1024 + n0 + (tid & 3) * 16, 0, 1);
    async_wait0();                       // drain ASYNCcnt before the barrier
    __syncthreads();
#pragma unroll
    for (int kk = 0; kk < 2; ++kk) {
      const int co = kk * 32;
      v16h a0 = load_frag16(At + (wm * 32) * 72 + co, 72);
      v16h a1 = load_frag16(At + (wm * 32 + 16) * 72 + co, 72);
      v16h b0 = load_frag16(Bt + (wn * 32) * 72 + co, 72);
      v16h b1 = load_frag16(Bt + (wn * 32 + 16) * 72 + co, 72);
      acc[0][0] = wmma_f16(a0, b0, acc[0][0]);
      acc[0][1] = wmma_f16(a0, b1, acc[0][1]);
      acc[1][0] = wmma_f16(a1, b0, acc[1][0]);
      acc[1][1] = wmma_f16(a1, b1, acc[1][1]);
    }
    __syncthreads();
  }
#pragma unroll
  for (int mm = 0; mm < 2; ++mm)
#pragma unroll
    for (int nn = 0; nn < 2; ++nn)
#pragma unroll
      for (int r = 0; r < 8; ++r) {
        int m = m0 + wm * 32 + mm * 16 + rowoff + r;
        int n = n0 + wn * 32 + nn * 16 + col;
        out[(long)m * 1024 + n] = acc[mm][nn][r] + bo[n];
      }
}

// ---------------------------------------------------------------------------
extern "C" void kernel_launch(void* const* d_in, const int* in_sizes, int n_in,
                              void* d_out, int out_size, void* d_ws, size_t ws_size,
                              hipStream_t stream) {
  (void)in_sizes; (void)n_in; (void)out_size; (void)ws_size;
  const float* q     = (const float*)d_in[0];
  const float* k     = (const float*)d_in[1];
  const float* v     = (const float*)d_in[2];
  const float* Wq    = (const float*)d_in[3];
  const float* bq    = (const float*)d_in[4];
  const float* Wk    = (const float*)d_in[5];
  const float* bk    = (const float*)d_in[6];
  const float* Wv    = (const float*)d_in[7];
  const float* bv    = (const float*)d_in[8];
  const float* Wo    = (const float*)d_in[9];
  const float* bo    = (const float*)d_in[10];
  const float* tw    = (const float*)d_in[11];
  const float* gamma = (const float*)d_in[12];
  const float* beta  = (const float*)d_in[13];
  float* wsf = (float*)d_ws;
  float* out = (float*)d_out;

  prep_kernel<<<dim3(256), dim3(256), 0, stream>>>(tw, wsf);
  qkv_gemm_kernel<<<dim3(16, 64, 3), dim3(256), 0, stream>>>(q, k, v, Wq, Wk, Wv,
                                                             bq, bk, bv, wsf);
  combine_kernel<<<dim3(16, 128, 3), dim3(256), 0, stream>>>(wsf);
  attention_kernel<<<dim3(16, 128), dim3(128), 0, stream>>>(wsf);
  ln_kernel<<<dim3(8192), dim3(256), 0, stream>>>(gamma, beta, wsf);
  out_gemm_kernel<<<dim3(16, 64), dim3(256), 0, stream>>>(Wo, bo, wsf, out);
}
